// sn_lstm_52716428591761
// MI455X (gfx1250) — matrix operand order repaired and checked by reading, not compiled, not run on hardware
//
#include <hip/hip_runtime.h>

// ---------------------------------------------------------------------------
// SN-LSTM persistent WMMA kernel for gfx1250 (MI455X).
//   B=16 batches -> 16 workgroups (batches independent: spatial shift is
//   intra-batch). 256 threads = 8 wave32 -> wave w owns row tile mt=w of the
//   S=128 rows. A-activations + h staging in LDS (312KB of 320KB/WGP),
//   c-state in VGPRs, weights packed as f16 WMMA fragments in workspace
//   (L2 resident, ~0.75MB).
// ---------------------------------------------------------------------------

#define B_   16
#define T_   128
#define S_   128
#define TP_  20
#define TOUT 107          // T-1-TP
#define KN_  224          // n/s cell K: 32 (x pad) + 3*64 (h,ha,hb)
#define KD_  512          // d cell K: 128 (hn|hs) + 3*128 (hd,hda,hdb)
#define NN_  256          // 4*NH
#define ND_  512          // 4*DH
#define NH_  64
#define DH_  128

typedef __attribute__((ext_vector_type(16))) _Float16 v16h;
typedef __attribute__((ext_vector_type(8)))  _Float16 v8h;
typedef __attribute__((ext_vector_type(8)))  float    v8f;

// Scheduling fence between gate groups: allow VALU(2)|SALU(4) to cross, pin
// VMEM/DS/WMMA so the next group's fragment loads are not hoisted across the
// previous group's epilogue (which caused scratch spills of live B-frags).
#if defined(__has_builtin)
#if __has_builtin(__builtin_amdgcn_sched_barrier)
#define SCHED_FENCE() __builtin_amdgcn_sched_barrier(6)
#endif
#endif
#ifndef SCHED_FENCE
#define SCHED_FENCE()
#endif

__device__ __forceinline__ float sigmoidf_(float x) {
  return 1.0f / (1.0f + __expf(-x));
}
__device__ __forceinline__ float tanhf_(float x) {
  return 2.0f / (1.0f + __expf(-2.0f * x)) - 1.0f;
}

union AFrag { v16h v; v8h h[2]; };

// ---------------------------------------------------------------------------
// Weight pack: f32 [Wx;pad;Wh;Wa;Wb] -> f16 WMMA B fragments (32x16 per frag).
// Fragment f = kt*Ntiles + nt, element e = lane*16 + j with
//   col = nt*16 + (lane&15),  k = kt*32 + (lane>>4)*8 + (j&7) + (j>>3)*16
// matching the per-lane v16h load in the GEMM (ISA 7.12.2 B layout).
// ---------------------------------------------------------------------------
__global__ void pack_kernel(const float* __restrict__ Wx,
                            const float* __restrict__ Wh,
                            const float* __restrict__ Wa,
                            const float* __restrict__ Wb,
                            _Float16* __restrict__ dst,
                            int xrows, int koff, int hdim, int K, int N) {
  const int Ntiles = N >> 4;
  const int total = K * N;
  for (int idx = blockIdx.x * blockDim.x + threadIdx.x; idx < total;
       idx += gridDim.x * blockDim.x) {
    const int f = idx >> 9;          // /512 elements per fragment
    const int e = idx & 511;
    const int kt = f / Ntiles, nt = f % Ntiles;
    const int lane = e >> 4, j = e & 15;
    const int n = nt * 16 + (lane & 15);
    const int k = kt * 32 + (lane >> 4) * 8 + (j & 7) + (j >> 3) * 16;
    float v = 0.0f;
    if (k < xrows) {
      v = Wx[k * N + n];
    } else if (k >= koff) {
      const int kk = k - koff;
      const int seg = kk / hdim;
      const int kr = kk % hdim;
      const float* W = (seg == 0) ? Wh : ((seg == 1) ? Wa : Wb);
      v = W[kr * N + n];
    }
    dst[idx] = (_Float16)v;
  }
}

// ---------------------------------------------------------------------------
// One LSTM cell step for row tile mt: GEMM over K (KT tiles of 32) producing
// gate groups; i/f/u/o tiles for the same columns share lane/VGPR slots, so
// the nonlinearity is in-register. First K-tile peeled (inline-0 src2);
// K loop unrolled x2 (double-buffered registers). h goes straight to an LDS
// staging buffer (row-major S_ x H) -- wave-private region, read only after
// the following barrier -- so no register stash survives across phases.
// ---------------------------------------------------------------------------
template <int KT, int NG>
__device__ __forceinline__ void cell_step(const _Float16* __restrict__ Ab, int Kst,
                                          const _Float16* __restrict__ Wp,
                                          const float* __restrict__ bias,
                                          v8f* cstate,
                                          _Float16* __restrict__ Hstg,
                                          int mt, int lane) {
  const int Ntiles = NG * 4;
  const int H = NG * 16;
  const int r = lane & 15;
  const int hf = lane >> 4;
  const _Float16* arow = Ab + (mt * 16 + r) * Kst + 8 * hf;
#pragma unroll
  for (int g = 0; g < NG; ++g) {
    SCHED_FENCE();
    const _Float16* ap = arow;
    const _Float16* wp = Wp + (size_t)g * 512 + (size_t)lane * 16;

    // ---- peeled kt = 0: accumulate onto inline zero ----
    AFrag a0;
    a0.h[0] = *(const v8h*)(ap);
    a0.h[1] = *(const v8h*)(ap + 16);
    v16h bi0 = *(const v16h*)(wp);
    v16h bf0 = *(const v16h*)(wp + (size_t)NG * 512);
    v16h bu0 = *(const v16h*)(wp + (size_t)(2 * NG) * 512);
    v16h bo0 = *(const v16h*)(wp + (size_t)(3 * NG) * 512);
    v8f ai = __builtin_amdgcn_wmma_f32_16x16x32_f16(false, a0.v, false, bi0, (short)0, (v8f){}, false, false);
    v8f af = __builtin_amdgcn_wmma_f32_16x16x32_f16(false, a0.v, false, bf0, (short)0, (v8f){}, false, false);
    v8f au = __builtin_amdgcn_wmma_f32_16x16x32_f16(false, a0.v, false, bu0, (short)0, (v8f){}, false, false);
    v8f ao = __builtin_amdgcn_wmma_f32_16x16x32_f16(false, a0.v, false, bo0, (short)0, (v8f){}, false, false);
    ap += 32;
    wp += (size_t)Ntiles * 512;

#pragma unroll 2
    for (int kt = 1; kt < KT; ++kt) {
      AFrag a;
      a.h[0] = *(const v8h*)(ap);        // ds_load_b128
      a.h[1] = *(const v8h*)(ap + 16);   // ds_load_b128
      v16h bi = *(const v16h*)(wp);
      v16h bf = *(const v16h*)(wp + (size_t)NG * 512);
      v16h bu = *(const v16h*)(wp + (size_t)(2 * NG) * 512);
      v16h bo = *(const v16h*)(wp + (size_t)(3 * NG) * 512);
      ai = __builtin_amdgcn_wmma_f32_16x16x32_f16(false, a.v, false, bi, (short)0, ai, false, false);
      af = __builtin_amdgcn_wmma_f32_16x16x32_f16(false, a.v, false, bf, (short)0, af, false, false);
      au = __builtin_amdgcn_wmma_f32_16x16x32_f16(false, a.v, false, bu, (short)0, au, false, false);
      ao = __builtin_amdgcn_wmma_f32_16x16x32_f16(false, a.v, false, bo, (short)0, ao, false, false);
      ap += 32;
      wp += (size_t)Ntiles * 512;
    }
    const int col = g * 16 + r;
    const float bbi = bias[col];
    const float bbf = bias[H + col];
    const float bbu = bias[2 * H + col];
    const float bbo = bias[3 * H + col];
    // nonlinearity + write h straight to staging (accum layout: element e is
    // row mt*16 + 8*hf + e, column col)
    _Float16* hout = Hstg + (mt * 16 + 8 * hf) * H + col;
#pragma unroll
    for (int e = 0; e < 8; ++e) {
      float iv = sigmoidf_(ai[e] + bbi);
      float fv = sigmoidf_(af[e] + bbf);
      float uv = tanhf_(au[e] + bbu);
      float ov = sigmoidf_(ao[e] + bbo);
      float c2 = fv * cstate[g][e] + iv * uv;
      cstate[g][e] = c2;
      hout[e * H] = (_Float16)(ov * tanhf_(c2));   // ds_store_b16
    }
  }
}

// ---------------------------------------------------------------------------
// Publish staging -> A-buffer slots with the +-1 spatial shift, vectorized in
// 8-half (16B) chunks:
//   A[row][baseS+c]   = h[row][c]      (Wh operand)
//   A[row-1][baseA+c] = h[row][c]      (ha[s]=h[s+1])
//   A[row+1][baseB+c] = h[row][c]      (hb[s]=h[s-1])
//   A2[row][base2+c]  = h[row][c]      (d-cell dec_in slot)
// Boundary slots stay zero (never written after the one-time LDS clear).
// ---------------------------------------------------------------------------
template <int H>
__device__ __forceinline__ void publish(const _Float16* __restrict__ Hstg,
                                        _Float16* __restrict__ A, int Kst,
                                        int baseS, int baseA, int baseB,
                                        _Float16* __restrict__ A2, int Kst2, int base2,
                                        int tid) {
  const int CH = H / 8;                 // 16B chunks per row
#pragma unroll 2
  for (int u = tid; u < S_ * CH; u += 256) {
    const int row = u / CH;
    const int cc = (u % CH) * 8;
    const v8h v = *(const v8h*)(Hstg + row * H + cc);
    *(v8h*)(A + row * Kst + baseS + cc) = v;
    if (row >= 1)      *(v8h*)(A + (row - 1) * Kst + baseA + cc) = v;
    if (row <= S_ - 2) *(v8h*)(A + (row + 1) * Kst + baseB + cc) = v;
    if (A2 != nullptr) *(v8h*)(A2 + row * Kst2 + base2 + cc) = v;
  }
}

// ---------------------------------------------------------------------------
// Persistent per-batch recurrence kernel.
// ---------------------------------------------------------------------------
__global__ __launch_bounds__(256, 1) void snlstm_main(
    const float* __restrict__ xin,
    const float* __restrict__ n_b, const float* __restrict__ s_b,
    const float* __restrict__ d_b,
    const float* __restrict__ on_W, const float* __restrict__ on_b,
    const float* __restrict__ os_W, const float* __restrict__ os_b,
    const _Float16* __restrict__ Wn, const _Float16* __restrict__ Ws,
    const _Float16* __restrict__ Wd,
    float* __restrict__ out) {
  extern __shared__ _Float16 smem[];
  _Float16* An   = smem;                     // 128 x 224 f16
  _Float16* As   = An + S_ * KN_;            // 128 x 224 f16
  _Float16* Ad   = As + S_ * KN_;            // 128 x 512 f16
  _Float16* stgN = Ad + S_ * KD_;            // 128 x 64  f16
  _Float16* stgS = stgN + S_ * NH_;          // 128 x 64  f16
  _Float16* stgD = stgS + S_ * NH_;          // 128 x 128 f16
  float* outbuf  = (float*)(stgD + S_ * DH_);// 2 x 128 f32

  const int b = blockIdx.x;
  const int tid = threadIdx.x;
  const int lane = tid & 31;
  const int wave = tid >> 5;

  // one-time zero of activation LDS (defines h(-1)=0 and boundary zeros)
  const int totalH = S_ * KN_ * 2 + S_ * KD_;
  for (int i = tid; i < totalH; i += 256) smem[i] = (_Float16)0.0f;
  for (int i = tid; i < 2 * S_; i += 256) outbuf[i] = 0.0f;

  v8f cN[4], cS[4], cD[8];
#pragma unroll
  for (int i = 0; i < 4; ++i) { cN[i] = (v8f){}; cS[i] = (v8f){}; }
#pragma unroll
  for (int i = 0; i < 8; ++i) cD[i] = (v8f){};

  // x_0 into the x-slots of A_n / A_s
  if (tid < S_) {
    const float4 x0 = *(const float4*)(xin + (((size_t)b * T_) * S_ + tid) * 4);
    An[tid * KN_ + 0] = (_Float16)x0.x;
    An[tid * KN_ + 1] = (_Float16)x0.y;
    An[tid * KN_ + 2] = (_Float16)x0.z;
    As[tid * KN_ + 0] = (_Float16)x0.w;
  }
  __syncthreads();

#pragma unroll 1
  for (int t = 0; t < T_ - 1; ++t) {
    // --- n-cell & s-cell GEMMs (read A_n / A_s; h -> staging) ---
    cell_step<7, 4>(An, KN_, Wn, n_b, cN, stgN, wave, lane);
    cell_step<7, 4>(As, KN_, Ws, s_b, cS, stgS, wave, lane);
    __syncthreads();
    // --- publish h_n, h_s (A_n/A_s recurrence slots + A_d dec_in) ---
    publish<NH_>(stgN, An, KN_, 32, 96, 160, Ad, KD_, 0,  tid);
    publish<NH_>(stgS, As, KN_, 32, 96, 160, Ad, KD_, 64, tid);
    __syncthreads();
    // --- d-cell GEMM (reads A_d; h -> staging) ---
    cell_step<16, 8>(Ad, KD_, Wd, d_b, cD, stgD, wave, lane);
    __syncthreads();
    // --- publish h_d (its own recurrence slots in A_d) ---
    publish<DH_>(stgD, Ad, KD_, 128, 256, 384, (_Float16*)nullptr, 0, 0, tid);
    // (stgD itself remains valid for the output head below)
    __syncthreads();
    // --- per-row output heads: out0 = hd.on_W + on_b ; out1 = hd.os_W + os_b
    if (tid < S_) {
      float o0 = on_b[0], o1 = os_b[0];
      const v8h* hd8 = (const v8h*)(stgD + tid * DH_);
#pragma unroll 4
      for (int jj = 0; jj < DH_ / 8; ++jj) {
        const v8h hv = hd8[jj];            // ds_load_b128
#pragma unroll
        for (int e = 0; e < 8; ++e) {
          const float hf32 = (float)hv[e];
          o0 += hf32 * on_W[jj * 8 + e];
          o1 += hf32 * os_W[jj * 8 + e];
        }
      }
      outbuf[tid] = o0;
      outbuf[S_ + tid] = o1;
    }
    __syncthreads();
    // --- _next_input assembly + refill x_{t+1} slots for next iteration ---
    if (tid < S_) {
      const int s = tid;
      const float4 x1 = *(const float4*)(xin + (((size_t)b * T_ + (t + 1)) * S_ + s) * 4);
      An[s * KN_ + 0] = (_Float16)x1.x;
      An[s * KN_ + 1] = (_Float16)x1.y;
      An[s * KN_ + 2] = (_Float16)x1.z;
      As[s * KN_ + 0] = (_Float16)x1.w;
      if (t >= TP_) {
        const float out0 = outbuf[s];
        const float out1 = outbuf[S_ + s];
        float inflow, spd;
        if (s == 0) {
          const float4 x10 = *(const float4*)(xin + (((size_t)b * T_ + (t + 1)) * S_) * 4);
          inflow = x10.y;   // next_x[:,0,1]
          spd    = x10.w;   // next_x[:,0,3]
        } else {
          inflow = outbuf[s - 1];  // output[:, s-1, 0]
          spd    = out1;           // output[:, s, 1]
        }
        const float xt2 = xin[(((size_t)b * T_ + t) * S_ + s) * 4 + 2];
        const float numcalc = xt2 + inflow - out0;
        float4 res;
        res.x = out0; res.y = inflow; res.z = numcalc; res.w = spd;
        *(float4*)(out + (((size_t)b * TOUT + (t - TP_)) * S_ + s) * 4) = res;
      }
    }
    __syncthreads();
  }
}

// ---------------------------------------------------------------------------
extern "C" void kernel_launch(void* const* d_in, const int* in_sizes, int n_in,
                              void* d_out, int out_size, void* d_ws, size_t ws_size,
                              hipStream_t stream) {
  (void)in_sizes; (void)n_in; (void)out_size; (void)ws_size;

  const float* xin  = (const float*)d_in[0];
  const float* n_Wx = (const float*)d_in[1];
  const float* n_Wh = (const float*)d_in[2];
  const float* n_Wa = (const float*)d_in[3];
  const float* n_Wb = (const float*)d_in[4];
  const float* n_b  = (const float*)d_in[5];
  const float* s_Wx = (const float*)d_in[6];
  const float* s_Wh = (const float*)d_in[7];
  const float* s_Wa = (const float*)d_in[8];
  const float* s_Wb = (const float*)d_in[9];
  const float* s_b  = (const float*)d_in[10];
  const float* d_Wx = (const float*)d_in[11];
  const float* d_Wh = (const float*)d_in[12];
  const float* d_Wa = (const float*)d_in[13];
  const float* d_Wb = (const float*)d_in[14];
  const float* d_b  = (const float*)d_in[15];
  const float* on_W = (const float*)d_in[16];
  const float* on_b = (const float*)d_in[17];
  const float* os_W = (const float*)d_in[18];
  const float* os_b = (const float*)d_in[19];

  _Float16* pN = (_Float16*)d_ws;                 // 224*256 = 57344 halfs
  _Float16* pS = pN + (size_t)KN_ * NN_;          // 57344 halfs
  _Float16* pD = pS + (size_t)KN_ * NN_;          // 512*512 = 262144 halfs

  const int totN = KN_ * NN_;
  const int totD = KD_ * ND_;
  pack_kernel<<<(totN + 255) / 256, 256, 0, stream>>>(n_Wx, n_Wh, n_Wa, n_Wb, pN, 3, 32, 64, KN_, NN_);
  pack_kernel<<<(totN + 255) / 256, 256, 0, stream>>>(s_Wx, s_Wh, s_Wa, s_Wb, pS, 1, 32, 64, KN_, NN_);
  pack_kernel<<<(totD + 255) / 256, 256, 0, stream>>>(d_Wx, d_Wh, d_Wa, d_Wb, pD, 128, 128, 128, KD_, ND_);

  const size_t lds_bytes =
      (size_t)(S_ * KN_ * 2 + S_ * KD_ + 2 * S_ * NH_ + S_ * DH_) * sizeof(_Float16) +
      (size_t)2 * S_ * sizeof(float);
  snlstm_main<<<dim3(B_), dim3(256), lds_bytes, stream>>>(
      xin, n_b, s_b, d_b, on_W, on_b, os_W, os_b, pN, pS, pD, (float*)d_out);
}
